// SpeciesIndexNet_85435489452600
// MI455X (gfx1250) — matrix operand order repaired and checked by reading, not compiled, not run on hardware
//
#include <hip/hip_runtime.h>

typedef __attribute__((ext_vector_type(16))) _Float16 v16h;
typedef __attribute__((ext_vector_type(8)))  _Float16 v8h;
typedef __attribute__((ext_vector_type(4)))  _Float16 v4h;
typedef __attribute__((ext_vector_type(8)))  float    v8f;
typedef __attribute__((ext_vector_type(4)))  float    v4f;

#define NSPEC    4
#define GROUP_SZ 50000
#define D_EMB    512
#define D_HID    512
#define D_OUT    256
#define M_TILE   64
#define LDH      520   // padded LDS row stride in halves (1040 B, 16B-aligned)

__device__ __forceinline__ float silu_f(float v) {
  return v * (1.0f / (1.0f + __expf(-v)));
}

// A-fragment (16x32 f16, wave32) from a row-major f16 row in LDS.
// Lane (l&15) = M-row; halves 0..15 hold K = 16*khalf+0..15.
__device__ __forceinline__ v16h load_a_h16(const _Float16* p) {
  v8h a0 = *(const v8h*)(p);
  v8h a1 = *(const v8h*)(p + 8);
  v16h a;
#pragma unroll
  for (int j = 0; j < 8; ++j) { a[j] = a0[j]; a[8 + j] = a1[j]; }
  return a;
}

// Transpose + fp32->f16 convert: W[s][k][n] -> Wt[s][n][k]
__global__ void transpose_w_kernel(const float* __restrict__ W,
                                   _Float16* __restrict__ Wt,
                                   int K, int N, int total) {
  int t = blockIdx.x * blockDim.x + threadIdx.x;
  if (t >= total) return;
  int k  = t % K;
  int rn = t / K;
  int n  = rn % N;
  int sp = rn / N;
  Wt[t] = (_Float16)W[((size_t)sp * K + k) * N + n];
}

__global__ __launch_bounds__(256) void species_mlp_kernel(
    const float* __restrict__ emb,    // [N_ATOMS, 512] fp32
    const int*   __restrict__ idx,    // [S, GROUP]
    const float* __restrict__ b1,     // [S, 512]
    const float* __restrict__ b2,     // [S, 512]
    const float* __restrict__ b3,     // [S, 256]
    const _Float16* __restrict__ w1t, // [S][512][512]  (N-major, K contiguous)
    const _Float16* __restrict__ w2t, // [S][512][512]
    const _Float16* __restrict__ w3t, // [S][256][512]
    float* __restrict__ out)          // [N_ATOMS, 256]
{
  __shared__ __align__(16) _Float16 hbuf[M_TILE * LDH];
  __shared__ int atomIds[M_TILE];

  const int s      = blockIdx.y;
  const int gblock = blockIdx.x * M_TILE;
  const int tid    = threadIdx.x;
  const int wave   = tid >> 5;      // 0..7 : disjoint 64-col N chunk (layers 1/2)
  const int lane   = tid & 31;
  const int mrow   = lane & 15;
  const int khalf  = lane >> 4;     // 0/1
  const int ncol   = lane & 15;

  // ---- prologue: cache atom ids; stage gathered x tile into LDS as f16 ----
  if (tid < M_TILE) {
    int g = gblock + tid;
    if (g >= GROUP_SZ) g = GROUP_SZ - 1;
    atomIds[tid] = idx[s * GROUP_SZ + g];
  }
#pragma unroll 1
  for (int i = tid; i < M_TILE * (D_EMB / 4); i += 256) {
    const int row = i >> 7;          // D_EMB/4 = 128 float4 per row
    const int c4  = i & 127;
    int g = gblock + row;
    if (g >= GROUP_SZ) g = GROUP_SZ - 1;
    const int atom = idx[s * GROUP_SZ + g];
    const v4f x = *(const v4f*)(emb + (size_t)atom * D_EMB + c4 * 4);
    v4h h;
#pragma unroll
    for (int j = 0; j < 4; ++j) h[j] = (_Float16)x[j];
    *(v4h*)&hbuf[row * LDH + c4 * 4] = h;
  }
  __syncthreads();

  const v8f vzero = {0.f, 0.f, 0.f, 0.f, 0.f, 0.f, 0.f, 0.f};
  const int nbase = wave * 64;

  // ---------------- layer 1: silu(x @ W1 + b1) -> hbuf ----------------
  {
    v8f acc[4][4];
#pragma unroll
    for (int ms = 0; ms < 4; ++ms)
#pragma unroll
      for (int nt = 0; nt < 4; ++nt) acc[ms][nt] = vzero;
    const _Float16* w = w1t + (size_t)s * D_HID * D_EMB;
#pragma unroll 1
    for (int kt = 0; kt < D_EMB / 32; ++kt) {
      v16h a[4];
#pragma unroll
      for (int ms = 0; ms < 4; ++ms)
        a[ms] = load_a_h16(&hbuf[(ms * 16 + mrow) * LDH + kt * 32 + khalf * 16]);
#pragma unroll
      for (int nt = 0; nt < 4; ++nt) {
        const int n = nbase + nt * 16 + ncol;
        const v16h b = *(const v16h*)(w + (size_t)n * D_EMB + kt * 32 + khalf * 16);
#pragma unroll
        for (int ms = 0; ms < 4; ++ms)
          acc[ms][nt] = __builtin_amdgcn_wmma_f32_16x16x32_f16(
              false, a[ms], false, b, (short)0, acc[ms][nt], false, false);
      }
    }
    __syncthreads();   // all x reads done before overwriting with h1
#pragma unroll
    for (int nt = 0; nt < 4; ++nt) {
      const int n = nbase + nt * 16 + ncol;
      const float bias = b1[s * D_HID + n];
#pragma unroll
      for (int ms = 0; ms < 4; ++ms)
#pragma unroll
        for (int r = 0; r < 8; ++r) {
          const int m = ms * 16 + r + 8 * khalf;   // D layout: VGPR r -> M = r + 8*khalf
          hbuf[m * LDH + n] = (_Float16)silu_f(acc[ms][nt][r] + bias);
        }
    }
  }
  __syncthreads();

  // ---------------- layer 2: silu(h1 @ W2 + b2) -> hbuf (reuse) ----------------
  {
    v8f acc[4][4];
#pragma unroll
    for (int ms = 0; ms < 4; ++ms)
#pragma unroll
      for (int nt = 0; nt < 4; ++nt) acc[ms][nt] = vzero;
    const _Float16* w = w2t + (size_t)s * D_HID * D_HID;
#pragma unroll 1
    for (int kt = 0; kt < D_HID / 32; ++kt) {
      v16h a[4];
#pragma unroll
      for (int ms = 0; ms < 4; ++ms)
        a[ms] = load_a_h16(&hbuf[(ms * 16 + mrow) * LDH + kt * 32 + khalf * 16]);
#pragma unroll
      for (int nt = 0; nt < 4; ++nt) {
        const int n = nbase + nt * 16 + ncol;
        const v16h b = *(const v16h*)(w + (size_t)n * D_HID + kt * 32 + khalf * 16);
#pragma unroll
        for (int ms = 0; ms < 4; ++ms)
          acc[ms][nt] = __builtin_amdgcn_wmma_f32_16x16x32_f16(
              false, a[ms], false, b, (short)0, acc[ms][nt], false, false);
      }
    }
    __syncthreads();   // all h1 reads done before overwriting with h2
#pragma unroll
    for (int nt = 0; nt < 4; ++nt) {
      const int n = nbase + nt * 16 + ncol;
      const float bias = b2[s * D_HID + n];
#pragma unroll
      for (int ms = 0; ms < 4; ++ms)
#pragma unroll
        for (int r = 0; r < 8; ++r) {
          const int m = ms * 16 + r + 8 * khalf;
          hbuf[m * LDH + n] = (_Float16)silu_f(acc[ms][nt][r] + bias);
        }
    }
  }
  __syncthreads();

  // ---------------- layer 3: h2 @ W3 + b3 -> scatter to out[idx] ----------------
  {
    v8f acc[4][2];
#pragma unroll
    for (int ms = 0; ms < 4; ++ms)
#pragma unroll
      for (int nt = 0; nt < 2; ++nt) acc[ms][nt] = vzero;
    const _Float16* w = w3t + (size_t)s * D_OUT * D_HID;
    const int nbase3 = wave * 32;    // 8 waves x 32 cols = 256
#pragma unroll 1
    for (int kt = 0; kt < D_HID / 32; ++kt) {
      v16h a[4];
#pragma unroll
      for (int ms = 0; ms < 4; ++ms)
        a[ms] = load_a_h16(&hbuf[(ms * 16 + mrow) * LDH + kt * 32 + khalf * 16]);
#pragma unroll
      for (int nt = 0; nt < 2; ++nt) {
        const int n = nbase3 + nt * 16 + ncol;
        const v16h b = *(const v16h*)(w + (size_t)n * D_HID + kt * 32 + khalf * 16);
#pragma unroll
        for (int ms = 0; ms < 4; ++ms)
          acc[ms][nt] = __builtin_amdgcn_wmma_f32_16x16x32_f16(
              false, a[ms], false, b, (short)0, acc[ms][nt], false, false);
      }
    }
    int am[4][8];
#pragma unroll
    for (int ms = 0; ms < 4; ++ms)
#pragma unroll
      for (int r = 0; r < 8; ++r)
        am[ms][r] = atomIds[ms * 16 + r + 8 * khalf];
#pragma unroll
    for (int nt = 0; nt < 2; ++nt) {
      const int n = nbase3 + nt * 16 + ncol;
      const float bias = b3[s * D_OUT + n];
#pragma unroll
      for (int ms = 0; ms < 4; ++ms)
#pragma unroll
        for (int r = 0; r < 8; ++r)
          out[(size_t)am[ms][r] * D_OUT + n] = acc[ms][nt][r] + bias;
    }
  }
}

extern "C" void kernel_launch(void* const* d_in, const int* in_sizes, int n_in,
                              void* d_out, int out_size, void* d_ws, size_t ws_size,
                              hipStream_t stream) {
  (void)in_sizes; (void)n_in; (void)out_size; (void)ws_size;
  // setup_inputs order: species, embedding, idx, W1, b1, W2, b2, W3, b3
  const float* emb = (const float*)d_in[1];
  const int*   idx = (const int*)d_in[2];
  const float* W1  = (const float*)d_in[3];
  const float* b1  = (const float*)d_in[4];
  const float* W2  = (const float*)d_in[5];
  const float* b2  = (const float*)d_in[6];
  const float* W3  = (const float*)d_in[7];
  const float* b3  = (const float*)d_in[8];
  float* out = (float*)d_out;

  _Float16* w1t = (_Float16*)d_ws;
  _Float16* w2t = w1t + (size_t)NSPEC * D_HID * D_EMB;
  _Float16* w3t = w2t + (size_t)NSPEC * D_HID * D_HID;

  const int e1 = NSPEC * D_EMB * D_HID;   // 1,048,576
  const int e2 = NSPEC * D_HID * D_HID;   // 1,048,576
  const int e3 = NSPEC * D_HID * D_OUT;   //   524,288

  transpose_w_kernel<<<(e1 + 255) / 256, 256, 0, stream>>>(W1, w1t, D_EMB, D_HID, e1);
  transpose_w_kernel<<<(e2 + 255) / 256, 256, 0, stream>>>(W2, w2t, D_HID, D_HID, e2);
  transpose_w_kernel<<<(e3 + 255) / 256, 256, 0, stream>>>(W3, w3t, D_HID, D_OUT, e3);

  dim3 grid((GROUP_SZ + M_TILE - 1) / M_TILE, NSPEC);
  species_mlp_kernel<<<grid, 256, 0, stream>>>(emb, idx, b1, b2, b3,
                                               w1t, w2t, w3t, out);
}
